// PPNP_10557029614175
// MI455X (gfx1250) — compile-verified, matrix operand order checked by reading, not run on hardware
//
#include <hip/hip_runtime.h>
#include <hip/hip_bf16.h>

// ---------------------------------------------------------------------------
// PPNP: out = segment_sum(A_data * X[A_col], A_row),  X = relu(F@W1+b1)@W2+b2
// bf16 WMMA fused MLP (h lives in VGPRs), pre-swizzled fragment-major weights
// (coalesced b128 loads), bank-conflict-free LDS, sched_group_barrier-driven
// software pipelining (load i+1 overlaps WMMA i), atomic edge scatter.
// ---------------------------------------------------------------------------

typedef __attribute__((ext_vector_type(16))) __bf16 v16bf;
typedef __attribute__((ext_vector_type(8)))  float  v8f;

#define D_IN  512
#define D_HID 256
#define D_OUT 32

#define FEAT_STRIDE 68   // 64 dwords payload + 4 pad  -> conflict-free b128
#define H_STRIDE    72   // 64 halfword payload + 8 pad -> conflict-free b128

// sched_group_barrier masks (AMDGPU IGroupLP)
#define SG_MFMA    0x008   // matches isMFMAorWMMA -> v_wmma on gfx1250
#define SG_VMEM_RD 0x020
#define SG_DS_RD   0x100

union ABfrag { v16bf v; unsigned u[8]; uint4 q[2]; };

__device__ __forceinline__ unsigned short f2bf(float f) {
    unsigned u = __float_as_uint(f);
    unsigned r = u + 0x7FFFu + ((u >> 16) & 1u);   // round-to-nearest-even
    return (unsigned short)(r >> 16);
}
__device__ __forceinline__ unsigned pack2(float lo, float hi) {
    return (unsigned)f2bf(lo) | ((unsigned)f2bf(hi) << 16);
}
// bf16 WMMA fragment dword slot -> K-pair index within a 32-K step
__device__ __forceinline__ int pidx(int v, int khalf) {
    return (v < 4) ? (v + khalf * 4) : (8 + (v - 4) + khalf * 4);
}

// ---------------------------------------------------------------------------
// Kernel 1: pre-swizzle weights into exact B-fragment order so the GEMM loads
// them with fully coalesced global_load_b128:
//   W1f[((kstep*16 + jt)*32 + lane)*8 + v],  kstep=K/32 (16), jt=N-tile (16)
//   W2f[((ks2 *2  + ot)*32 + lane)*8 + v],   ks2 =K/32 (8),  ot=N-tile (2)
// ---------------------------------------------------------------------------
__global__ __launch_bounds__(256) void pack_weights(
    const float* __restrict__ W1, const float* __restrict__ W2,
    unsigned* __restrict__ W1f, unsigned* __restrict__ W2f) {
    int t = blockIdx.x * 256 + threadIdx.x;
    if (t < 16 * 16 * 32 * 8) {                    // 65536
        int v     = t & 7;
        int lane  = (t >> 3) & 31;
        int jt    = (t >> 8) & 15;
        int kstep = t >> 12;
        int j     = jt * 16 + (lane & 15);
        int kp    = kstep * 16 + pidx(v, lane >> 4);
        W1f[t] = pack2(W1[(2 * kp) * D_HID + j], W1[(2 * kp + 1) * D_HID + j]);
    }
    if (t < 8 * 2 * 32 * 8) {                      // 4096
        int v    = t & 7;
        int lane = (t >> 3) & 31;
        int ot   = (t >> 8) & 1;
        int ks2  = t >> 9;
        int o    = ot * 16 + (lane & 15);
        int kp   = ks2 * 16 + pidx(v, lane >> 4);
        W2f[t] = pack2(W2[(2 * kp) * D_OUT + o], W2[(2 * kp + 1) * D_OUT + o]);
    }
}

// ---------------------------------------------------------------------------
// Kernel 2: zero the output (scatter accumulates with atomics)
// ---------------------------------------------------------------------------
__global__ __launch_bounds__(256) void zero_out(float* __restrict__ out, int n) {
    int i = blockIdx.x * 256 + threadIdx.x;
    if (i < n) out[i] = 0.0f;
}

// ---------------------------------------------------------------------------
// Kernel 3: fused MLP.  Block = 256 threads = 8 wave32, M-tile = 128 rows.
// ---------------------------------------------------------------------------
__global__ __launch_bounds__(256) void mlp_fused(
    const float* __restrict__ feat, const float* __restrict__ b1,
    const float* __restrict__ b2, const unsigned* __restrict__ W1f,
    const unsigned* __restrict__ W2f, float* __restrict__ X, int nNodes) {

    __shared__ unsigned       sFeat[128][FEAT_STRIDE]; // 128 rows x 128 K bf16  (~34.8KB)
    __shared__ unsigned short sH[8][16][H_STRIDE];     // per-wave 16x64 h chunk (~18.4KB)

    const int tid   = threadIdx.x;
    const int lane  = tid & 31;
    const int wave  = tid >> 5;
    const int khalf = lane >> 4;
    const int lm    = lane & 15;
    const long gBase = (long)blockIdx.x * 128;

    const unsigned* sFrow = &sFeat[wave * 16 + lm][khalf * 4];

    // ---- h accumulators (16 N-tiles of 16x16 f32), init with bias b1 ----
    v8f hacc[16];
#pragma unroll
    for (int jt = 0; jt < 16; ++jt) {
        float bv = b1[jt * 16 + lm];
#pragma unroll
        for (int r = 0; r < 8; ++r) hacc[jt][r] = bv;
    }

    // ---- GEMM1: h += F(bf16) @ W1(bf16), K = 512 in four 128-chunks ----
    for (int kc = 0; kc < D_IN; kc += 128) {
        __syncthreads();
        // stage 128x128 fp32 -> bf16 pairs in LDS (16 float4 per thread)
#pragma unroll
        for (int i = 0; i < 16; ++i) {
            int idx = i * 256 + tid;           // 0..4095
            int row = idx >> 5;                // 0..127
            int c4  = idx & 31;                // float4 within chunk
            long grow = gBase + row;
            if (grow > (long)nNodes - 1) grow = (long)nNodes - 1;
            const float4 f = ((const float4*)(feat + grow * D_IN + kc))[c4];
            sFeat[row][c4 * 2 + 0] = pack2(f.x, f.y);
            sFeat[row][c4 * 2 + 1] = pack2(f.z, f.w);
        }
        __syncthreads();

        const int kstep0 = kc >> 5;
        ABfrag afrag[2], bfrag[2];
        // pipeline prologue: fragment 0
        afrag[0].q[0] = *(const uint4*)(sFrow + 0);
        afrag[0].q[1] = *(const uint4*)(sFrow + 8);
        {
            const uint4* wp = (const uint4*)(W1f + (((kstep0 * 16) * 32 + lane) << 3));
            bfrag[0].q[0] = wp[0];
            bfrag[0].q[1] = wp[1];
        }
        // steady state: issue loads for fragment it+1, WMMA on fragment it.
        // sched_group_barrier pins the [2 VMEM][2 DS][1 WMMA] pipeline shape so
        // RA must keep two live fragment buffers (no full loadcnt drains).
#pragma unroll
        for (int it = 0; it < 64; ++it) {
            const int s  = it >> 4;            // k-step within chunk (0..3)
            const int jt = it & 15;            // N-tile
            const int nit = it + 1;
            bool aload = false;
            if (nit < 64) {
                const int ns  = nit >> 4;
                const int njt = nit & 15;
                if (njt == 0) {
                    afrag[ns & 1].q[0] = *(const uint4*)(sFrow + ns * 16);
                    afrag[ns & 1].q[1] = *(const uint4*)(sFrow + ns * 16 + 8);
                    aload = true;
                }
                const uint4* wp = (const uint4*)(
                    W1f + ((((kstep0 + ns) * 16 + njt) * 32 + lane) << 3));
                bfrag[nit & 1].q[0] = wp[0];
                bfrag[nit & 1].q[1] = wp[1];
            }
            hacc[jt] = __builtin_amdgcn_wmma_f32_16x16x32_bf16(
                false, afrag[s & 1].v, false, bfrag[it & 1].v,
                (short)0, hacc[jt], false, false);
            if (nit < 64) __builtin_amdgcn_sched_group_barrier(SG_VMEM_RD, 2, 0);
            if (aload)    __builtin_amdgcn_sched_group_barrier(SG_DS_RD, 2, 0);
            __builtin_amdgcn_sched_group_barrier(SG_MFMA, 1, 0);
        }
    }

    // ---- GEMM2: X = relu(h)(bf16) @ W2(bf16) + b2, K = 256 in 64-chunks ----
    v8f xacc[2];
#pragma unroll
    for (int ot = 0; ot < 2; ++ot) {
        float bv = b2[ot * 16 + lm];
#pragma unroll
        for (int r = 0; r < 8; ++r) xacc[ot][r] = bv;
    }

#pragma unroll
    for (int jc = 0; jc < 4; ++jc) {
        // ReLU + C-layout -> A-layout re-staging through per-wave LDS region
#pragma unroll
        for (int q = 0; q < 4; ++q) {
            int jt = jc * 4 + q;
#pragma unroll
            for (int r = 0; r < 8; ++r) {
                float h = hacc[jt][r];
                h = h > 0.0f ? h : 0.0f;
                sH[wave][r + 8 * khalf][q * 16 + lm] = f2bf(h);
            }
        }
        asm volatile("s_wait_dscnt 0x0" ::: "memory");  // per-wave LDS RAW fence
#pragma unroll
        for (int k2 = 0; k2 < 64; k2 += 32) {
            const int ks2 = (jc * 64 + k2) >> 5;
            const unsigned* hrow = (const unsigned*)&sH[wave][lm][0];
            ABfrag a;
            a.q[0] = *(const uint4*)(hrow + k2 / 2 + khalf * 4);
            a.q[1] = *(const uint4*)(hrow + k2 / 2 + 8 + khalf * 4);
#pragma unroll
            for (int ot = 0; ot < 2; ++ot) {
                const uint4* wp = (const uint4*)(W2f + (((ks2 * 2 + ot) * 32 + lane) << 3));
                ABfrag b;
                b.q[0] = wp[0];
                b.q[1] = wp[1];
                xacc[ot] = __builtin_amdgcn_wmma_f32_16x16x32_bf16(
                    false, a.v, false, b.v, (short)0, xacc[ot], false, false);
            }
        }
        asm volatile("s_wait_dscnt 0x0" ::: "memory");  // before next chunk overwrite
    }

    // ---- store X tile (f32, C-layout -> row major) ----
#pragma unroll
    for (int ot = 0; ot < 2; ++ot)
#pragma unroll
        for (int r = 0; r < 8; ++r) {
            long row = gBase + wave * 16 + r + 8 * khalf;
            if (row < nNodes) X[row * D_OUT + ot * 16 + lm] = xacc[ot][r];
        }
}

// ---------------------------------------------------------------------------
// Kernel 4: edge scatter. 8 lanes per edge: float4 gather of X[A_col] (L2-hot,
// 12.8MB << 192MB L2), scale by A_data, global_atomic_add_f32 into out[A_row].
// ---------------------------------------------------------------------------
__global__ __launch_bounds__(256) void scatter_edges(
    const float* __restrict__ X, const float* __restrict__ Ad,
    const int* __restrict__ Arow, const int* __restrict__ Acol,
    float* __restrict__ out, int nE) {
    long t = (long)blockIdx.x * 256 + threadIdx.x;
    long e = t >> 3;
    int  q = (int)(t & 7);
    if (e >= nE) return;
    float a = Ad[e];
    long  c = Acol[e];
    long  r = Arow[e];
    const float4 x = ((const float4*)(X + c * D_OUT))[q];
    float* o = out + r * D_OUT + q * 4;
    atomicAdd(o + 0, a * x.x);
    atomicAdd(o + 1, a * x.y);
    atomicAdd(o + 2, a * x.z);
    atomicAdd(o + 3, a * x.w);
}

// ---------------------------------------------------------------------------
extern "C" void kernel_launch(void* const* d_in, const int* in_sizes, int n_in,
                              void* d_out, int out_size, void* d_ws, size_t ws_size,
                              hipStream_t stream) {
    const float* feat = (const float*)d_in[0];
    const float* Ad   = (const float*)d_in[1];
    const float* W1   = (const float*)d_in[2];
    const float* b1   = (const float*)d_in[3];
    const float* W2   = (const float*)d_in[4];
    const float* b2   = (const float*)d_in[5];
    const int*   Arow = (const int*)d_in[6];
    const int*   Acol = (const int*)d_in[7];
    float* out = (float*)d_out;

    const int nNodes = in_sizes[0] / D_IN;     // 100000
    const int nE     = in_sizes[1];            // 1600000

    // workspace layout
    unsigned* W1f = (unsigned*)d_ws;                       // 65536 u32 = 256KB
    unsigned* W2f = W1f + 16 * 16 * 32 * 8;                // 4096 u32  = 16KB
    float*    X   = (float*)(W2f + 8 * 2 * 32 * 8);        // nNodes*32 f32

    pack_weights<<<256, 256, 0, stream>>>(W1, W2, W1f, W2f);
    zero_out<<<(out_size + 255) / 256, 256, 0, stream>>>(out, out_size);
    mlp_fused<<<(nNodes + 127) / 128, 256, 0, stream>>>(feat, b1, b2, W1f, W2f, X, nNodes);
    {
        long threads = (long)nE * 8;
        int  blocks  = (int)((threads + 255) / 256);
        scatter_edges<<<blocks, 256, 0, stream>>>(X, Ad, Arow, Acol, out, nE);
    }
}
